// PiT_71382356460016
// MI455X (gfx1250) — compile-verified, matrix operand order checked by reading, not run on hardware
//
#include <hip/hip_runtime.h>
#include <hip/hip_bf16.h>

// ---------------------------------------------------------------------------
// PiT (position-attention transformer) for gfx1250 / MI455X.
// All matmuls run on v_wmma_f32_16x16x32_f16 (f16 in, f32 accumulate).
// Attention probabilities are batch-independent -> computed once per head,
// stored f16 to halve HBM traffic on the bandwidth-dominant tensors.
// Each wave computes a 16x32 tile (one A fragment feeds two WMMAs) with a
// branch-free double-buffered K loop (K % 64 == 0 for every GEMM here).
// Softmax row staging uses async global->LDS copies (ASYNCcnt).
// ---------------------------------------------------------------------------

typedef __attribute__((ext_vector_type(16))) _Float16 v16h;
typedef __attribute__((ext_vector_type(8)))  float    v8f;
typedef __attribute__((ext_vector_type(4)))  int      i32x4_t;

#define B_   8
#define NF_  4096
#define NL_  1024
#define C_   256
#define H_   8
#define VD_  32

#ifndef __has_builtin
#define __has_builtin(x) 0
#endif
#if __has_builtin(__builtin_amdgcn_global_load_async_to_lds_b128)
#define HAS_ASYNC_LDS 1
typedef __attribute__((address_space(1))) i32x4_t gbl_i32x4;
typedef __attribute__((address_space(3))) i32x4_t lds_i32x4;
#else
#define HAS_ASYNC_LDS 0
#endif

__device__ __forceinline__ float gelu_tanh(float x) {
    const float c = 0.7978845608028654f;   // sqrt(2/pi)
    float t = tanhf(c * (x + 0.044715f * x * x * x));
    return 0.5f * x * (1.0f + t);
}

union FragU { v16h v; unsigned int u[8]; };

// Fragment packing per CDNA5 ISA 7.12.2 (16-bit 16x32 A layout; B symmetric):
// lane: idx = lane&15 (M row for A / N col for B), half = lane>>4
// vgpr v: kbase = (v>>2)*16 + half*8 + (v&3)*2 -> one dword = 2 contig f16.
__device__ __forceinline__ void load_frag(FragU& f, const _Float16* __restrict__ rowp,
                                          int k0, int half) {
#pragma unroll
    for (int v = 0; v < 8; ++v) {
        const int kb = k0 + ((v >> 2) << 4) + (half << 3) + ((v & 3) << 1);
        f.u[v] = *(const unsigned int*)(rowp + kb);
    }
}

#define WMMA_F16(ACC, AF, BF) \
    ACC = __builtin_amdgcn_wmma_f32_16x16x32_f16(false, (AF).v, false, (BF).v, \
                                                 (short)0, ACC, false, false)

// One 16x32 output tile: A row-major (16 x K), BT0/BT1 row-major (16 x K each).
// Double-buffered without register rotation: set1 loads overlap set0 WMMAs and
// vice versa. Requires K % 64 == 0 and K >= 64 (true for all GEMMs here).
__device__ __forceinline__ void wmma_tile16x32(const _Float16* __restrict__ A, long lda,
                                               const _Float16* __restrict__ BT0,
                                               const _Float16* __restrict__ BT1, long ldbt,
                                               int K, int lane, v8f& acc0, v8f& acc1) {
    const int idx  = lane & 15;
    const int half = lane >> 4;
    const _Float16* arow  = A   + (long)idx * lda;
    const _Float16* b0row = BT0 + (long)idx * ldbt;
    const _Float16* b1row = BT1 + (long)idx * ldbt;
    FragU a0, p0, q0, a1, p1, q1;
    load_frag(a0, arow, 0, half);
    load_frag(p0, b0row, 0, half);
    load_frag(q0, b1row, 0, half);
    int k0 = 0;
    for (; k0 < K - 64; k0 += 64) {
        __builtin_prefetch(arow + k0 + 128, 0, 1);      // -> global_prefetch_b8
        __builtin_prefetch(b0row + k0 + 128, 0, 1);
        load_frag(a1, arow, k0 + 32, half);
        load_frag(p1, b0row, k0 + 32, half);
        load_frag(q1, b1row, k0 + 32, half);
        WMMA_F16(acc0, a0, p0);
        WMMA_F16(acc1, a0, q0);
        load_frag(a0, arow, k0 + 64, half);
        load_frag(p0, b0row, k0 + 64, half);
        load_frag(q0, b1row, k0 + 64, half);
        WMMA_F16(acc0, a1, p1);
        WMMA_F16(acc1, a1, q1);
    }
    // peeled tail: exactly two k-steps remain (k0 == K - 64)
    load_frag(a1, arow, k0 + 32, half);
    load_frag(p1, b0row, k0 + 32, half);
    load_frag(q1, b1row, k0 + 32, half);
    WMMA_F16(acc0, a0, p0);
    WMMA_F16(acc1, a0, q0);
    WMMA_F16(acc0, a1, p1);
    WMMA_F16(acc1, a1, q1);
}

// ---------------------------------------------------------------------------
// Encoder: h = gelu(x @ W[4,256] + b), out f16 [B, NF, 256]
// ---------------------------------------------------------------------------
__global__ void encoder_kernel(const float* __restrict__ x, const float* __restrict__ w,
                               const float* __restrict__ bias, _Float16* __restrict__ out,
                               int total) {
    int i = blockIdx.x * blockDim.x + threadIdx.x;
    if (i >= total) return;
    int c = i & (C_ - 1);
    long bn = (long)(i >> 8);
    const float* xp = x + bn * 4;
    float acc = bias[c];
#pragma unroll
    for (int j = 0; j < 4; ++j) acc += xp[j] * w[j * C_ + c];
    out[i] = (_Float16)gelu_tanh(acc);
}

// ---------------------------------------------------------------------------
// Stabilized softmax over -m_dist*scale, one block per (head, query-row).
// scale = tan(0.25*pi*(1-1e-7)*(1+sin(r_h))) >= 0, so max(-sd) = -scale*min(m).
// Row staged in LDS via async global->LDS copies when available.
// Writes normalized probabilities f16 to att[H, Nq, Nk].
// ---------------------------------------------------------------------------
__global__ void softmax_att_kernel(const float* __restrict__ m_dist,
                                   const float* __restrict__ r,
                                   _Float16* __restrict__ att, int Nq, int Nk) {
    extern __shared__ float row[];              // Nk floats
    __shared__ float red[8];
    const int q = blockIdx.x, h = blockIdx.y;
    const int tid = threadIdx.x, lane = tid & 31, wave = tid >> 5;
    const int nwave = blockDim.x >> 5;
    const float scale = tanf(0.78539816339744831f * (1.0f - 1e-7f) * (1.0f + sinf(r[h])));
    const float* src = m_dist + ((long)h * Nq + q) * (long)Nk;

    float mn = 3.4e38f;
#if HAS_ASYNC_LDS
    // 16B per lane per op; Nk*4 is a multiple of 16 and of blockDim.x*16.
    for (int i = tid * 4; i < Nk; i += blockDim.x * 4) {
        __builtin_amdgcn_global_load_async_to_lds_b128(
            (gbl_i32x4*)(src + i), (lds_i32x4*)(row + i), 0, 0);
    }
#if __has_builtin(__builtin_amdgcn_s_wait_asynccnt)
    __builtin_amdgcn_s_wait_asynccnt(0);
#else
    asm volatile("s_wait_asynccnt 0" ::: "memory");
#endif
    __syncthreads();
    for (int i = tid; i < Nk; i += blockDim.x) mn = fminf(mn, row[i]);
#else
    for (int i = tid; i < Nk; i += blockDim.x) { float v = src[i]; row[i] = v; mn = fminf(mn, v); }
#endif
    for (int off = 16; off; off >>= 1) mn = fminf(mn, __shfl_xor(mn, off, 32));
    if (lane == 0) red[wave] = mn;
    __syncthreads();
    if (tid == 0) { float m2 = red[0]; for (int i = 1; i < nwave; ++i) m2 = fminf(m2, red[i]); red[0] = m2; }
    __syncthreads();
    mn = red[0];
    __syncthreads();

    float s = 0.f;
    for (int i = tid; i < Nk; i += blockDim.x) {
        float e = expf(scale * (mn - row[i]));
        row[i] = e; s += e;
    }
    for (int off = 16; off; off >>= 1) s += __shfl_xor(s, off, 32);
    if (lane == 0) red[wave] = s;
    __syncthreads();
    if (tid == 0) { float t = 0.f; for (int i = 0; i < nwave; ++i) t += red[i]; red[0] = t; }
    __syncthreads();
    const float inv = 1.0f / red[0];

    _Float16* dst = att + ((long)h * Nq + q) * (long)Nk;
    for (int i = tid; i < Nk; i += blockDim.x) dst[i] = (_Float16)(row[i] * inv);
}

// ---------------------------------------------------------------------------
// Generic batched WMMA GEMM:  D[z] = act(A[z](MxK) @ B(KxN) + bias + add[z])
// B supplied transposed (BT: N x K f16). 4 waves/block, one 16x32 tile per
// wave (tile predication is wave-uniform -> EXEC all-ones around WMMA).
// ---------------------------------------------------------------------------
__global__ void gemm_dense_kernel(const _Float16* __restrict__ A, long a_bs, long lda,
                                  const _Float16* __restrict__ BT, long ldbt,
                                  const float* __restrict__ bias,
                                  const float* __restrict__ addsrc, long add_bs, long add_ld,
                                  float* __restrict__ out32, long o32_bs, long o32_rs, long o32_cs,
                                  _Float16* __restrict__ out16, long o16_bs, long o16_rs, long o16_cs,
                                  int M, int N, int K, int act) {
    const int lane = threadIdx.x & 31;
    const int wave = threadIdx.x >> 5;
    const int tiles_n = N >> 5;
    const int tile = blockIdx.x * 4 + wave;
    if (tile >= (M >> 4) * tiles_n) return;
    const int tm = (tile / tiles_n) << 4;
    const int tn = (tile % tiles_n) << 5;
    const long z = blockIdx.y;

    v8f acc0 = {0.f,0.f,0.f,0.f,0.f,0.f,0.f,0.f};
    v8f acc1 = {0.f,0.f,0.f,0.f,0.f,0.f,0.f,0.f};
    wmma_tile16x32(A + z * a_bs + (long)tm * lda, lda,
                   BT + (long)tn * ldbt, BT + (long)(tn + 16) * ldbt, ldbt,
                   K, lane, acc0, acc1);

    const int half = lane >> 4;
    const int n0 = tn + (lane & 15);
    const int n1 = n0 + 16;
    const float b0 = bias ? bias[n0] : 0.0f;
    const float b1 = bias ? bias[n1] : 0.0f;
#pragma unroll
    for (int v = 0; v < 8; ++v) {
        const int m = tm + v + (half << 3);
        float r0 = acc0[v] + b0;
        float r1 = acc1[v] + b1;
        if (addsrc) {
            const float* ap = addsrc + z * add_bs + (long)m * add_ld;
            r0 += ap[n0]; r1 += ap[n1];
        }
        if (act) { r0 = gelu_tanh(r0); r1 = gelu_tanh(r1); }
        if (out32) {
            float* op = out32 + z * o32_bs + (long)m * o32_rs;
            op[(long)n0 * o32_cs] = r0; op[(long)n1 * o32_cs] = r1;
        }
        if (out16) {
            _Float16* op = out16 + z * o16_bs + (long)m * o16_rs;
            op[(long)n0 * o16_cs] = (_Float16)r0; op[(long)n1 * o16_cs] = (_Float16)r1;
        }
    }
}

// ---------------------------------------------------------------------------
// Attention apply:  out[b, :, h*32 : h*32+32] = gelu(att[h] @ value_t[b,h]^T)
// att: [H, Nq, Nk] f16 (A).  value_t: [B, 256, Nk] f16 (transposed ->
// contiguous-K B fragments).  One 16x32 tile (full head) per wave.
// ---------------------------------------------------------------------------
__global__ void att_apply_kernel(const _Float16* __restrict__ att,
                                 const _Float16* __restrict__ value_t,
                                 _Float16* __restrict__ out16, int Nq, int Nk) {
    const int lane = threadIdx.x & 31;
    const int wave = threadIdx.x >> 5;
    const int tile = blockIdx.x * 4 + wave;          // Nq/16 row-tiles
    if (tile >= (Nq >> 4)) return;
    const int tm = tile << 4;
    const int b = blockIdx.y, h = blockIdx.z;

    const _Float16* Ab = att + ((long)h * Nq + tm) * (long)Nk;
    const _Float16* B0 = value_t + ((long)b * C_ + h * VD_) * (long)Nk;
    v8f acc0 = {0.f,0.f,0.f,0.f,0.f,0.f,0.f,0.f};
    v8f acc1 = {0.f,0.f,0.f,0.f,0.f,0.f,0.f,0.f};
    wmma_tile16x32(Ab, Nk, B0, B0 + (long)16 * Nk, Nk, Nk, lane, acc0, acc1);

    const int half = lane >> 4;
    const int n0 = h * VD_ + (lane & 15);
#pragma unroll
    for (int v = 0; v < 8; ++v) {
        const int m = tm + v + (half << 3);
        _Float16* op = out16 + ((long)b * Nq + m) * C_;
        op[n0]      = (_Float16)gelu_tanh(acc0[v]);
        op[n0 + 16] = (_Float16)gelu_tanh(acc1[v]);
    }
}

// ---------------------------------------------------------------------------
// Weight prep: f32 [K,N] -> f16 transposed [N,K];  head weights [H,C,VD] ->
// concatenated transposed [H*VD, C].
// ---------------------------------------------------------------------------
__global__ void w_transpose_f16(const float* __restrict__ W, _Float16* __restrict__ WT,
                                int K, int N) {
    int i = blockIdx.x * blockDim.x + threadIdx.x;
    if (i >= K * N) return;
    int k = i / N, n = i % N;
    WT[(long)n * K + k] = (_Float16)W[i];
}

__global__ void headw_to_wcatT(const float* __restrict__ w, _Float16* __restrict__ WT) {
    int i = blockIdx.x * blockDim.x + threadIdx.x;     // H*C*VD = 65536
    if (i >= H_ * C_ * VD_) return;
    int h = i / (C_ * VD_), rem = i % (C_ * VD_);
    int c = rem / VD_, k = rem % VD_;
    WT[((long)(h * VD_ + k)) * C_ + c] = (_Float16)w[i];
}

// Final N=1 projection (VALU): out = h @ w[256,1] + b
__global__ void de_fc2_kernel(const _Float16* __restrict__ h, const float* __restrict__ w,
                              const float* __restrict__ bias, float* __restrict__ out, int total) {
    int i = blockIdx.x * blockDim.x + threadIdx.x;
    if (i >= total) return;
    const _Float16* row = h + (long)i * C_;
    float acc = bias[0];
#pragma unroll 8
    for (int c = 0; c < C_; ++c) acc += (float)row[c] * w[c];
    out[i] = acc;
}

// ---------------------------------------------------------------------------
extern "C" void kernel_launch(void* const* d_in, const int* in_sizes, int n_in,
                              void* d_out, int out_size, void* d_ws, size_t ws_size,
                              hipStream_t stream) {
    (void)in_sizes; (void)n_in; (void)out_size; (void)ws_size;
    const float* x           = (const float*)d_in[0];
    const float* m_down      = (const float*)d_in[1];
    const float* m_p[2]      = {(const float*)d_in[2], (const float*)d_in[3]};
    const float* m_up        = (const float*)d_in[4];
    const float* en_w        = (const float*)d_in[5];
    const float* en_b        = (const float*)d_in[6];
    const float* down_r      = (const float*)d_in[7];
    const float* down_w      = (const float*)d_in[8];
    const float* pa_r[2]     = {(const float*)d_in[9],  (const float*)d_in[17]};
    const float* pa_w[2]     = {(const float*)d_in[10], (const float*)d_in[18]};
    const float* f1w[2]      = {(const float*)d_in[11], (const float*)d_in[19]};
    const float* f1b[2]      = {(const float*)d_in[12], (const float*)d_in[20]};
    const float* f2w[2]      = {(const float*)d_in[13], (const float*)d_in[21]};
    const float* f2b[2]      = {(const float*)d_in[14], (const float*)d_in[22]};
    const float* rsw[2]      = {(const float*)d_in[15], (const float*)d_in[23]};
    const float* rsb[2]      = {(const float*)d_in[16], (const float*)d_in[24]};
    const float* up_r        = (const float*)d_in[25];
    const float* up_w        = (const float*)d_in[26];
    const float* de1_w       = (const float*)d_in[27];
    const float* de1_b       = (const float*)d_in[28];
    const float* de2_w       = (const float*)d_in[29];
    const float* de2_b       = (const float*)d_in[30];
    float* out = (float*)d_out;

    // ---- workspace layout ----
    char* ws = (char*)d_ws;
    size_t o = 0;
    auto alloc = [&](size_t bytes) { size_t r = o; o = (o + bytes + 255) & ~(size_t)255; return r; };
    _Float16* att     = (_Float16*)(ws + alloc((size_t)H_ * NL_ * NF_ * 2));   // 64 MiB, reused
    _Float16* value_t = (_Float16*)(ws + alloc((size_t)B_ * C_ * NF_ * 2));
    _Float16* h_full  = (_Float16*)(ws + alloc((size_t)B_ * NF_ * C_ * 2));
    _Float16* h_latA  = (_Float16*)(ws + alloc((size_t)B_ * NL_ * C_ * 2));
    _Float16* h_latB  = (_Float16*)(ws + alloc((size_t)B_ * NL_ * C_ * 2));
    _Float16* pa_buf  = (_Float16*)(ws + alloc((size_t)B_ * NL_ * C_ * 2));
    _Float16* mlp1    = (_Float16*)(ws + alloc((size_t)B_ * NL_ * C_ * 2));
    float*    mlp2    = (float*)   (ws + alloc((size_t)B_ * NL_ * C_ * 4));
    _Float16* h_full2 = (_Float16*)(ws + alloc((size_t)B_ * NF_ * C_ * 2));
    _Float16* de1_act = (_Float16*)(ws + alloc((size_t)B_ * NF_ * C_ * 2));
    const size_t WB = (size_t)C_ * C_ * 2;
    _Float16* wT_down = (_Float16*)(ws + alloc(WB));
    _Float16* wT_pa[2]= {(_Float16*)(ws + alloc(WB)), (_Float16*)(ws + alloc(WB))};
    _Float16* wT_up   = (_Float16*)(ws + alloc(WB));
    _Float16* wT_f1[2]= {(_Float16*)(ws + alloc(WB)), (_Float16*)(ws + alloc(WB))};
    _Float16* wT_f2[2]= {(_Float16*)(ws + alloc(WB)), (_Float16*)(ws + alloc(WB))};
    _Float16* wT_rs[2]= {(_Float16*)(ws + alloc(WB)), (_Float16*)(ws + alloc(WB))};
    _Float16* wT_de1  = (_Float16*)(ws + alloc(WB));

    // ---- weight prep ----
    headw_to_wcatT<<<dim3(256), 256, 0, stream>>>(down_w, wT_down);
    headw_to_wcatT<<<dim3(256), 256, 0, stream>>>(pa_w[0], wT_pa[0]);
    headw_to_wcatT<<<dim3(256), 256, 0, stream>>>(pa_w[1], wT_pa[1]);
    headw_to_wcatT<<<dim3(256), 256, 0, stream>>>(up_w, wT_up);
    for (int i = 0; i < 2; ++i) {
        w_transpose_f16<<<dim3(256), 256, 0, stream>>>(f1w[i], wT_f1[i], C_, C_);
        w_transpose_f16<<<dim3(256), 256, 0, stream>>>(f2w[i], wT_f2[i], C_, C_);
        w_transpose_f16<<<dim3(256), 256, 0, stream>>>(rsw[i], wT_rs[i], C_, C_);
    }
    w_transpose_f16<<<dim3(256), 256, 0, stream>>>(de1_w, wT_de1, C_, C_);

    // ---- encoder ----
    encoder_kernel<<<dim3((B_ * NF_ * C_) / 256), 256, 0, stream>>>(x, en_w, en_b, h_full, B_ * NF_ * C_);

    auto run_gemm = [&](const _Float16* A, long a_bs, long lda, const _Float16* BT,
                        const float* bias, const float* add, long add_bs,
                        float* o32, long o32_bs, long o32_rs, long o32_cs,
                        _Float16* o16, long o16_bs, long o16_rs, long o16_cs,
                        int M, int N, int K, int act, int batch) {
        int tiles = (M >> 4) * (N >> 5);
        gemm_dense_kernel<<<dim3((tiles + 3) / 4, batch), 128, 0, stream>>>(
            A, a_bs, lda, BT, (long)K, bias, add, add_bs, (long)C_,
            o32, o32_bs, o32_rs, o32_cs, o16, o16_bs, o16_rs, o16_cs, M, N, K, act);
    };
    auto run_att = [&](const _Float16* attp, const _Float16* vt, _Float16* o16,
                       int Nq, int Nk) {
        att_apply_kernel<<<dim3(((Nq >> 4) + 3) / 4, B_, H_), 128, 0, stream>>>(attp, vt, o16, Nq, Nk);
    };

    // ---- down attention: [B,4096,256] -> [B,1024,256] ----
    softmax_att_kernel<<<dim3(NL_, H_), 256, NF_ * 4, stream>>>(m_down, down_r, att, NL_, NF_);
    run_gemm(h_full, (long)NF_ * C_, C_, wT_down, nullptr, nullptr, 0,
             nullptr, 0, 0, 0, value_t, (long)C_ * NF_, 1, NF_, NF_, C_, C_, 0, B_);
    run_att(att, value_t, h_latA, NL_, NF_);

    // ---- two latent blocks ----
    _Float16* hcur = h_latA;
    _Float16* hnxt = h_latB;
    for (int i = 0; i < 2; ++i) {
        softmax_att_kernel<<<dim3(NL_, H_), 256, NL_ * 4, stream>>>(m_p[i], pa_r[i], att, NL_, NL_);
        run_gemm(hcur, (long)NL_ * C_, C_, wT_pa[i], nullptr, nullptr, 0,
                 nullptr, 0, 0, 0, value_t, (long)C_ * NL_, 1, NL_, NL_, C_, C_, 0, B_);
        run_att(att, value_t, pa_buf, NL_, NL_);
        // mlp fc1 (gelu) -> mlp1 f16
        run_gemm(pa_buf, (long)NL_ * C_, C_, wT_f1[i], f1b[i], nullptr, 0,
                 nullptr, 0, 0, 0, mlp1, (long)NL_ * C_, C_, 1, NL_, C_, C_, 1, B_);
        // mlp fc2 -> mlp2 f32
        run_gemm(mlp1, (long)NL_ * C_, C_, wT_f2[i], f2b[i], nullptr, 0,
                 mlp2, (long)NL_ * C_, C_, 1, nullptr, 0, 0, 0, NL_, C_, C_, 0, B_);
        // h = gelu(mlp2 + h @ res_w + res_b) -> hnxt f16
        run_gemm(hcur, (long)NL_ * C_, C_, wT_rs[i], rsb[i], mlp2, (long)NL_ * C_,
                 nullptr, 0, 0, 0, hnxt, (long)NL_ * C_, C_, 1, NL_, C_, C_, 1, B_);
        _Float16* t = hcur; hcur = hnxt; hnxt = t;
    }

    // ---- up attention: [B,1024,256] -> [B,4096,256] ----
    softmax_att_kernel<<<dim3(NF_, H_), 256, NL_ * 4, stream>>>(m_up, up_r, att, NF_, NL_);
    run_gemm(hcur, (long)NL_ * C_, C_, wT_up, nullptr, nullptr, 0,
             nullptr, 0, 0, 0, value_t, (long)C_ * NL_, 1, NL_, NL_, C_, C_, 0, B_);
    run_att(att, value_t, h_full2, NF_, NL_);

    // ---- decoder ----
    run_gemm(h_full2, (long)NF_ * C_, C_, wT_de1, de1_b, nullptr, 0,
             nullptr, 0, 0, 0, de1_act, (long)NF_ * C_, C_, 1, NF_, C_, C_, 1, B_);
    de_fc2_kernel<<<dim3((B_ * NF_ + 255) / 256), 256, 0, stream>>>(de1_act, de2_w, de2_b, out, B_ * NF_);
}
